// GAT_75746043232799
// MI455X (gfx1250) — matrix operand order repaired and checked by reading, not compiled, not run on hardware
//
#include <hip/hip_runtime.h>
#include <hip/hip_bf16.h>

#define GN 8192
#define GD 512
#define GNHID 64
#define GNCLASS 16
#define GNHEADS 4
#define GALPHA 0.2f

typedef _Float16 half_t;
typedef __attribute__((ext_vector_type(16))) _Float16 v16h;
typedef __attribute__((ext_vector_type(8)))  _Float16 v8h;
typedef __attribute__((ext_vector_type(8)))  float    v8f;

__device__ inline v8f wmma_f16(v16h a, v16h b, v8f c) {
  // D = A(16x32 f16) * B(32x16 f16) + C(16x16 f32)
  return __builtin_amdgcn_wmma_f32_16x16x32_f16(false, a, false, b, (short)0, c,
                                                false, false);
}

// Fragment from two contiguous 8-half runs (two global_load_b128).
__device__ inline v16h ld_frag16(const half_t* plo, const half_t* phi) {
  v8h lo = *(const v8h*)plo;
  v8h hi = *(const v8h*)phi;
  v16h r;
#pragma unroll
  for (int i = 0; i < 8; ++i) { r[i] = lo[i]; r[8 + i] = hi[i]; }
  return r;
}

// A fragment from an f32 row: K runs [klo..klo+7] and [khi..khi+7].
__device__ inline v16h ld_a_f32(const float* row, int klo, int khi) {
  float lo[8], hi[8];
  ((float4*)lo)[0] = *(const float4*)(row + klo);
  ((float4*)lo)[1] = *(const float4*)(row + klo + 4);
  ((float4*)hi)[0] = *(const float4*)(row + khi);
  ((float4*)hi)[1] = *(const float4*)(row + khi + 4);
  v16h r;
#pragma unroll
  for (int i = 0; i < 8; ++i) { r[i] = (half_t)lo[i]; r[8 + i] = (half_t)hi[i]; }
  return r;
}

// ---------------- weight packing: transpose to [Fout][K] f16 ----------------
__global__ void gat_pack(const float* __restrict__ W_heads,
                         const float* __restrict__ W_out,
                         half_t* __restrict__ WT1, half_t* __restrict__ WoutT) {
  int idx = blockIdx.x * blockDim.x + threadIdx.x;
  if (idx < GNHEADS * GNHID * GD) {            // WT1[h][n][k] = W_heads[h][k][n]
    int h = idx / (GNHID * GD);
    int rem = idx % (GNHID * GD);
    int n = rem / GD, k = rem % GD;
    WT1[idx] = (half_t)W_heads[((size_t)h * GD + k) * GNHID + n];
  }
  if (idx < GNCLASS * (GNHEADS * GNHID)) {     // WoutT[n][k] = W_out[k][n]
    int n = idx / (GNHEADS * GNHID), k = idx % (GNHEADS * GNHID);
    WoutT[idx] = (half_t)W_out[(size_t)k * GNCLASS + n];
  }
}

// ------- GEMM1: Wh = emb @ W_head -> WhT (f16 [head][64][N]) + src/dst -------
__global__ __launch_bounds__(32)
void gat_gemm1(const float* __restrict__ emb, const half_t* __restrict__ WT1,
               const float* __restrict__ a_heads, half_t* __restrict__ WhT,
               float* __restrict__ src, float* __restrict__ dst) {
  const int head = blockIdx.y;
  const int i0 = blockIdx.x * 16;
  const int lane = threadIdx.x;
  const int M = lane & 15, hh = lane >> 4;
  const int klo = hh * 8, khi = 16 + hh * 8;
  const float* arow = emb + (size_t)(i0 + M) * GD;
  const half_t* wt = WT1 + (size_t)head * GNHID * GD;
  v8f acc[4] = {};
  for (int kc = 0; kc < GD; kc += 32) {
    v16h A = ld_a_f32(arow + kc, klo, khi);
    v16h Bf[4];
#pragma unroll
    for (int t = 0; t < 4; ++t) {
      const half_t* bp = wt + (size_t)(t * 16 + M) * GD + kc + hh * 8;
      Bf[t] = ld_frag16(bp, bp + 16);
    }
#pragma unroll
    for (int t = 0; t < 4; ++t) acc[t] = wmma_f16(A, Bf[t], acc[t]);
  }
#pragma unroll
  for (int t = 0; t < 4; ++t) {
    v8h o;
#pragma unroll
    for (int r = 0; r < 8; ++r) o[r] = (half_t)acc[t][r];
    *(v8h*)(WhT + (size_t)(head * GNHID + t * 16 + M) * GN + i0 + 8 * hh) = o;
  }
  const float* ah = a_heads + head * 2 * GNHID;
#pragma unroll
  for (int r = 0; r < 8; ++r) {
    float s = 0.f, d = 0.f;
#pragma unroll
    for (int t = 0; t < 4; ++t) {
      float w = acc[t][r];
      s += w * ah[t * 16 + M];
      d += w * ah[GNHID + t * 16 + M];
    }
#pragma unroll
    for (int off = 8; off; off >>= 1) {
      s += __shfl_xor(s, off, 16);
      d += __shfl_xor(d, off, 16);
    }
    if (M == 0) {
      int row = i0 + r + 8 * hh;
      src[head * GN + row] = s;
      dst[head * GN + row] = d;
    }
  }
}

// ---------------- max over each length-n vector ----------------
__global__ void gat_rowmax(const float* __restrict__ v, float* __restrict__ out,
                           int n) {
  __shared__ float sm[256];
  const float* p = v + (size_t)blockIdx.x * n;
  float m = -3.0e38f;
  for (int i = threadIdx.x; i < n; i += blockDim.x) m = fmaxf(m, p[i]);
  sm[threadIdx.x] = m;
  __syncthreads();
  for (int s = 128; s; s >>= 1) {
    if ((int)threadIdx.x < s)
      sm[threadIdx.x] = fmaxf(sm[threadIdx.x], sm[threadIdx.x + s]);
    __syncthreads();
  }
  if (threadIdx.x == 0) out[blockIdx.x] = sm[0];
}

// ------- flash layer 1: elu(softmax(mask(lrelu(src+dst))) @ Wh) -------
// 256 threads = 8 waves = 4 heads x 2 column-halves; LDS combine at end.
// adj tiles are double-buffered: next tile's loads issue before processing.
__global__ __launch_bounds__(256)
void gat_flash1(const float* __restrict__ adj, const half_t* __restrict__ WhT,
                const float* __restrict__ src, const float* __restrict__ dstv,
                const float* __restrict__ maxdst, half_t* __restrict__ h1) {
  __shared__ float lacc[GNHEADS * GNHID * 16];   // [head][n][row] f32
  __shared__ float lden[GNHEADS * 16];
  const int wave = threadIdx.x >> 5;
  const int head = wave & 3;
  const int chalf = wave >> 2;                   // column half
  const int lane = threadIdx.x & 31;
  const int i0 = blockIdx.x * 16;
  const int M = lane & 15, hh = lane >> 4;
  const int klo = hh * 8, khi = 16 + hh * 8;
  const float srcM = src[head * GN + i0 + M];
  float t0 = srcM + maxdst[head];
  const float mrow = t0 > 0.f ? t0 : GALPHA * t0;  // safe row-wise shift
  const float* arow = adj + (size_t)(i0 + M) * GN;
  const float* dh = dstv + head * GN;
  const half_t* wt = WhT + (size_t)head * GNHID * GN;
  v8f acc[4] = {};
  float dpart = 0.f;
  const int jbeg = chalf * (GN / 2), jend = jbeg + (GN / 2);

  auto ld_adj = [&](float* b, int j) {
    ((float4*)b)[0] = *(const float4*)(arow + j + klo);
    ((float4*)b)[1] = *(const float4*)(arow + j + klo + 4);
    ((float4*)b)[2] = *(const float4*)(arow + j + khi);
    ((float4*)b)[3] = *(const float4*)(arow + j + khi + 4);
  };
  auto process = [&](const float* av, int j0) {
    float amax = av[0];
#pragma unroll
    for (int e = 1; e < 16; ++e) amax = fmaxf(amax, av[e]);
    if (__ballot(amax > 0.f) == 0ull) return;      // wave-uniform tile skip
    v16h Bf[4];                                    // B first: overlap exp chain
#pragma unroll
    for (int t = 0; t < 4; ++t) {
      const half_t* bp = wt + (size_t)(t * 16 + M) * GN + j0 + hh * 8;
      Bf[t] = ld_frag16(bp, bp + 16);
    }
    float dv[16];
    ((float4*)dv)[0] = *(const float4*)(dh + j0 + klo);
    ((float4*)dv)[1] = *(const float4*)(dh + j0 + klo + 4);
    ((float4*)dv)[2] = *(const float4*)(dh + j0 + khi);
    ((float4*)dv)[3] = *(const float4*)(dh + j0 + khi + 4);
    v16h P;
    float ds = 0.f;
#pragma unroll
    for (int e = 0; e < 16; ++e) {
      float xv = srcM + dv[e];
      xv = xv > 0.f ? xv : GALPHA * xv;
      float p = (av[e] > 0.f) ? __expf(xv - mrow) : 0.f;
      ds += p;
      P[e] = (half_t)p;
    }
    dpart += ds;
#pragma unroll
    for (int t = 0; t < 4; ++t) acc[t] = wmma_f16(P, Bf[t], acc[t]);
  };

  float b0[16], b1[16];
  ld_adj(b0, jbeg);
  for (int j0 = jbeg; j0 < jend; j0 += 64) {       // 2-deep pipeline
    if (j0 + 512 < jend) __builtin_prefetch(arow + j0 + 512, 0, 1);
    ld_adj(b1, j0 + 32);
    process(b0, j0);
    ld_adj(b0, (j0 + 64 < jend) ? j0 + 64 : jbeg); // clamped preload
    process(b1, j0 + 32);
  }

  dpart += __shfl_xor(dpart, 16, 32);              // per-row denom (this half)
  if (chalf == 1) {                                // publish partials
#pragma unroll
    for (int t = 0; t < 4; ++t)
      *(v8f*)&lacc[((head * GNHID + t * 16 + M) * 16) + 8 * hh] = acc[t];
    if (lane < 16) lden[head * 16 + M] = dpart;
  }
  __syncthreads();
  if (chalf == 0) {
    float dtot = dpart + lden[head * 16 + M];
#pragma unroll
    for (int t = 0; t < 4; ++t) {
      v8f o = *(v8f*)&lacc[((head * GNHID + t * 16 + M) * 16) + 8 * hh];
#pragma unroll
      for (int r = 0; r < 8; ++r) acc[t][r] += o[r];
    }
#pragma unroll
    for (int r = 0; r < 8; ++r) {
      int Mc = r + 8 * hh;
      float dd = __shfl(dtot, Mc, 32);
#pragma unroll
      for (int t = 0; t < 4; ++t) {
        float v = acc[t][r] / dd;
        v = v > 0.f ? v : __expf(v) - 1.f;         // ELU
        h1[(size_t)(i0 + Mc) * (GNHEADS * GNHID) + head * GNHID + t * 16 + M] =
            (half_t)v;
      }
    }
  }
}

// ------- GEMM2: Wh2 = h1 @ W_out -> Wh2T (f16 [16][N]) + src2/dst2 -------
__global__ __launch_bounds__(32)
void gat_gemm2(const half_t* __restrict__ h1, const half_t* __restrict__ WoutT,
               const float* __restrict__ a_out, half_t* __restrict__ Wh2T,
               float* __restrict__ src2, float* __restrict__ dst2) {
  const int i0 = blockIdx.x * 16;
  const int lane = threadIdx.x;
  const int M = lane & 15, hh = lane >> 4;
  const half_t* arow = h1 + (size_t)(i0 + M) * (GNHEADS * GNHID);
  v8f acc = {};
  for (int kc = 0; kc < GNHEADS * GNHID; kc += 32) {
    v16h A = ld_frag16(arow + kc + hh * 8, arow + kc + 16 + hh * 8);
    const half_t* bp = WoutT + (size_t)M * (GNHEADS * GNHID) + kc + hh * 8;
    acc = wmma_f16(A, ld_frag16(bp, bp + 16), acc);
  }
  {
    v8h o;
#pragma unroll
    for (int r = 0; r < 8; ++r) o[r] = (half_t)acc[r];
    *(v8h*)(Wh2T + (size_t)M * GN + i0 + 8 * hh) = o;
  }
#pragma unroll
  for (int r = 0; r < 8; ++r) {
    float s = acc[r] * a_out[M];
    float d = acc[r] * a_out[GNCLASS + M];
#pragma unroll
    for (int off = 8; off; off >>= 1) {
      s += __shfl_xor(s, off, 16);
      d += __shfl_xor(d, off, 16);
    }
    if (M == 0) {
      int row = i0 + r + 8 * hh;
      src2[row] = s;
      dst2[row] = d;
    }
  }
}

// ------- flash layer 2 + ELU + row log_softmax -> d_out -------
// 128 threads = 4 waves = 4 column quarters; LDS combine at end.
__global__ __launch_bounds__(128)
void gat_flash2(const float* __restrict__ adj, const half_t* __restrict__ Wh2T,
                const float* __restrict__ src2, const float* __restrict__ dst2,
                const float* __restrict__ max2, float* __restrict__ out) {
  __shared__ float lacc[3 * GNCLASS * 16];       // [slot][n][row]
  __shared__ float lden[3 * 16];
  const int qc = threadIdx.x >> 5;               // column quarter
  const int lane = threadIdx.x & 31;
  const int i0 = blockIdx.x * 16;
  const int M = lane & 15, hh = lane >> 4;
  const int klo = hh * 8, khi = 16 + hh * 8;
  const float srcM = src2[i0 + M];
  float t0 = srcM + max2[0];
  const float mrow = t0 > 0.f ? t0 : GALPHA * t0;
  const float* arow = adj + (size_t)(i0 + M) * GN;
  v8f acc = {};
  float dpart = 0.f;
  const int jbeg = qc * (GN / 4), jend = jbeg + (GN / 4);

  auto ld_adj = [&](float* b, int j) {
    ((float4*)b)[0] = *(const float4*)(arow + j + klo);
    ((float4*)b)[1] = *(const float4*)(arow + j + klo + 4);
    ((float4*)b)[2] = *(const float4*)(arow + j + khi);
    ((float4*)b)[3] = *(const float4*)(arow + j + khi + 4);
  };
  auto process = [&](const float* av, int j0) {
    float amax = av[0];
#pragma unroll
    for (int e = 1; e < 16; ++e) amax = fmaxf(amax, av[e]);
    if (__ballot(amax > 0.f) == 0ull) return;
    const half_t* bp = Wh2T + (size_t)M * GN + j0 + hh * 8;
    v16h Bf = ld_frag16(bp, bp + 16);
    float dv[16];
    ((float4*)dv)[0] = *(const float4*)(dst2 + j0 + klo);
    ((float4*)dv)[1] = *(const float4*)(dst2 + j0 + klo + 4);
    ((float4*)dv)[2] = *(const float4*)(dst2 + j0 + khi);
    ((float4*)dv)[3] = *(const float4*)(dst2 + j0 + khi + 4);
    v16h P;
    float ds = 0.f;
#pragma unroll
    for (int e = 0; e < 16; ++e) {
      float xv = srcM + dv[e];
      xv = xv > 0.f ? xv : GALPHA * xv;
      float p = (av[e] > 0.f) ? __expf(xv - mrow) : 0.f;
      ds += p;
      P[e] = (half_t)p;
    }
    dpart += ds;
    acc = wmma_f16(P, Bf, acc);
  };

  float b0[16], b1[16];
  ld_adj(b0, jbeg);
  for (int j0 = jbeg; j0 < jend; j0 += 64) {       // 2-deep pipeline
    if (j0 + 512 < jend) __builtin_prefetch(arow + j0 + 512, 0, 1);
    ld_adj(b1, j0 + 32);
    process(b0, j0);
    ld_adj(b0, (j0 + 64 < jend) ? j0 + 64 : jbeg);
    process(b1, j0 + 32);
  }

  dpart += __shfl_xor(dpart, 16, 32);
  if (qc > 0) {
    *(v8f*)&lacc[(((qc - 1) * GNCLASS + M) * 16) + 8 * hh] = acc;
    if (lane < 16) lden[(qc - 1) * 16 + M] = dpart;
  }
  __syncthreads();
  if (qc == 0) {
    float dtot = dpart;
#pragma unroll
    for (int s = 0; s < 3; ++s) {
      v8f o = *(v8f*)&lacc[((s * GNCLASS + M) * 16) + 8 * hh];
#pragma unroll
      for (int r = 0; r < 8; ++r) acc[r] += o[r];
      dtot += lden[s * 16 + M];
    }
#pragma unroll
    for (int r = 0; r < 8; ++r) {
      int Mc = r + 8 * hh;
      float dd = __shfl(dtot, Mc, 32);
      float v = acc[r] / dd;
      v = v > 0.f ? v : __expf(v) - 1.f;           // outer ELU
      float mx = v;
#pragma unroll
      for (int off = 8; off; off >>= 1) mx = fmaxf(mx, __shfl_xor(mx, off, 16));
      float ex = __expf(v - mx);
#pragma unroll
      for (int off = 8; off; off >>= 1) ex += __shfl_xor(ex, off, 16);
      out[(size_t)(i0 + Mc) * GNCLASS + M] = v - mx - __logf(ex);
    }
  }
}

extern "C" void kernel_launch(void* const* d_in, const int* in_sizes, int n_in,
                              void* d_out, int out_size, void* d_ws,
                              size_t ws_size, hipStream_t stream) {
  (void)in_sizes; (void)n_in; (void)out_size; (void)ws_size;
  const float* adj     = (const float*)d_in[1];
  const float* emb     = (const float*)d_in[2];
  const float* W_heads = (const float*)d_in[3];
  const float* a_heads = (const float*)d_in[4];
  const float* W_out   = (const float*)d_in[5];
  const float* a_out   = (const float*)d_in[6];

  size_t off = 0;
  auto carve = [&](size_t bytes) {
    char* p = (char*)d_ws + off;
    off += (bytes + 255) & ~(size_t)255;
    return (void*)p;
  };
  half_t* WT1   = (half_t*)carve((size_t)GNHEADS * GNHID * GD * 2);
  half_t* WoutT = (half_t*)carve((size_t)GNCLASS * GNHEADS * GNHID * 2);
  half_t* WhT1  = (half_t*)carve((size_t)GNHEADS * GNHID * GN * 2);
  float*  src1  = (float*)carve((size_t)GNHEADS * GN * 4);
  float*  dst1  = (float*)carve((size_t)GNHEADS * GN * 4);
  float*  max1  = (float*)carve(256);
  half_t* h1    = (half_t*)carve((size_t)GN * GNHEADS * GNHID * 2);
  half_t* Wh2T  = (half_t*)carve((size_t)GNCLASS * GN * 2);
  float*  src2  = (float*)carve((size_t)GN * 4);
  float*  dst2  = (float*)carve((size_t)GN * 4);
  float*  max2  = (float*)carve(256);

  gat_pack<<<512, 256, 0, stream>>>(W_heads, W_out, WT1, WoutT);
  gat_gemm1<<<dim3(GN / 16, GNHEADS), 32, 0, stream>>>(emb, WT1, a_heads, WhT1,
                                                       src1, dst1);
  gat_rowmax<<<GNHEADS, 256, 0, stream>>>(dst1, max1, GN);
  gat_flash1<<<GN / 16, 256, 0, stream>>>(adj, WhT1, src1, dst1, max1, h1);
  gat_gemm2<<<GN / 16, 32, 0, stream>>>(h1, WoutT, a_out, Wh2T, src2, dst2);
  gat_rowmax<<<1, 256, 0, stream>>>(dst2, max2, GN);
  gat_flash2<<<GN / 16, 128, 0, stream>>>(adj, Wh2T, src2, dst2, max2,
                                          (float*)d_out);
}
